// LigerExperts_25288767439422
// MI455X (gfx1250) — compile-verified, matrix operand order checked by reading, not run on hardware
//
#include <hip/hip_runtime.h>
#include <hip/hip_bf16.h>
#include <stdint.h>

#define E_      8
#define H_      2048
#define I_      1408
#define S_      2048
#define K_      2
#define TWO_I   2816
#define NPAIR   4096            // S_*K_
#define ROWPAD  64              // expert segment padding (4 M-subtiles of 16)
#define ROWCAP  4608            // NPAIR + E_*ROWPAD
#define ROWTILES 72             // ROWCAP/64

typedef __attribute__((ext_vector_type(16))) __bf16 v16bf;
typedef __attribute__((ext_vector_type(8)))  float  v8f;

union Frag { uint32_t u[8]; v16bf v; };

// pack two fp32 -> two bf16 in one dword (lo = a, hi = b)
__device__ __forceinline__ uint32_t pkcvt(float a, float b) {
#if __has_builtin(__builtin_amdgcn_cvt_pk_bf16_f32)
    auto r = __builtin_amdgcn_cvt_pk_bf16_f32(a, b);
    return __builtin_bit_cast(uint32_t, r);
#else
    union { float f; uint32_t u; } va, vb; va.f = a; vb.f = b;
    uint32_t ua = va.u + 0x8000u, ub = vb.u + 0x8000u;   // round-half-up
    return (ua >> 16) | (ub & 0xFFFF0000u);
#endif
}

__device__ __forceinline__ uint16_t f2bfu(float f) {
    union { float f; uint32_t u; } v; v.f = f;
    uint32_t r = v.u + 0x7FFFu + ((v.u >> 16) & 1u);
    return (uint16_t)(r >> 16);
}

// 8 consecutive fp32 -> 4 packed bf16 dwords
__device__ __forceinline__ void cvt8u(const float* __restrict__ p, uint32_t* d) {
    const float4* q = (const float4*)p;
    float4 x0 = q[0], x1 = q[1];
    d[0] = pkcvt(x0.x, x0.y); d[1] = pkcvt(x0.z, x0.w);
    d[2] = pkcvt(x1.x, x1.y); d[3] = pkcvt(x1.z, x1.w);
}

// 8 consecutive bf16 (16B) -> 4 dwords, single b128 load
__device__ __forceinline__ void ld8u(const unsigned short* __restrict__ p, uint32_t* d) {
    uint4 t = *(const uint4*)p;
    d[0] = t.x; d[1] = t.y; d[2] = t.z; d[3] = t.w;
}

// ---------------- bulk fp32 -> bf16 conversion (bandwidth-bound) ----------------

__global__ void moe_cvt(const float* __restrict__ src, unsigned short* __restrict__ dst, int n8) {
    int g = blockIdx.x * blockDim.x + threadIdx.x;
    if (g < n8) {
        const float4* q = (const float4*)src + (size_t)g * 2;
        float4 a = q[0], b = q[1];
        uint4 o;
        o.x = pkcvt(a.x, a.y); o.y = pkcvt(a.z, a.w);
        o.z = pkcvt(b.x, b.y); o.w = pkcvt(b.z, b.w);
        ((uint4*)dst)[g] = o;
    }
}

// ---------------- routing kernels ----------------

__global__ void moe_zero(float* __restrict__ out, int* __restrict__ meta,
                         unsigned int* __restrict__ xz) {
    int gid = blockIdx.x * blockDim.x + threadIdx.x;
    if (gid < S_ * H_) out[gid] = 0.f;
    if (gid < 16) meta[gid] = 0;          // cnt[8] + cursor[8]
    if (gid < H_ / 2) xz[gid] = 0;        // zero bf16 row for padding tokens
}

__global__ void moe_count(const int* __restrict__ idx, int* __restrict__ cnt) {
    int p = blockIdx.x * blockDim.x + threadIdx.x;
    if (p < NPAIR) atomicAdd(&cnt[idx[p] & (E_ - 1)], 1);
}

__global__ void moe_offsets(const int* __restrict__ cnt, int* __restrict__ off,
                            int* __restrict__ rowExpert, int* __restrict__ tokenId,
                            float* __restrict__ wpair) {
    __shared__ int soff[E_ + 1];
    if (threadIdx.x == 0) {
        int o = 0; soff[0] = 0; off[0] = 0;
        for (int e = 0; e < E_; ++e) {
            o += ((cnt[e] + ROWPAD - 1) / ROWPAD) * ROWPAD;
            soff[e+1] = o; off[e+1] = o;
        }
    }
    __syncthreads();
    for (int r = threadIdx.x; r < ROWTILES; r += blockDim.x) {
        int b = r * 64, ex = 0;
        for (int q = 1; q <= E_; ++q) if (b >= soff[q]) ex = q;
        rowExpert[r] = (ex >= E_) ? 0 : ex;
    }
    for (int p = threadIdx.x; p < ROWCAP; p += blockDim.x) {
        tokenId[p] = -1; wpair[p] = 0.f;
    }
}

__global__ void moe_scatter(const int* __restrict__ idx, const float* __restrict__ wts,
                            int* __restrict__ cursor, const int* __restrict__ off,
                            int* __restrict__ tokenId, float* __restrict__ wpair) {
    int p = blockIdx.x * blockDim.x + threadIdx.x;
    if (p < NPAIR) {
        int e = idx[p] & (E_ - 1);
        int pos = atomicAdd(&cursor[e], 1);
        int slot = off[e] + pos;
        tokenId[slot] = p / K_;
        wpair[slot] = wts[p];
    }
}

// ---------------- phase 1: h = silu(x@Wg^T) * (x@Wu^T), bf16 out ----------------
// grid = (ROWTILES, I/128), block = 256 (8 waves); wave = gate+up 16-col tile, 4 M-subtiles

template<bool PRE>
__global__ void moe_gateup(const float* __restrict__ x32, const unsigned short* __restrict__ xbf,
                           const float* __restrict__ g32, const unsigned short* __restrict__ gbf,
                           const int* __restrict__ rowExpert, const int* __restrict__ tokenId,
                           unsigned short* __restrict__ hbuf) {
    const int wave = threadIdx.x >> 5;
    const int lane = threadIdx.x & 31;
    const int rt   = blockIdx.x;
    const int e    = rowExpert[rt];
    const int hs   = (lane < 16) ? 0 : 8;   // K half-select / M offset per ISA layout
    const int lidx = lane & 15;

    int tok[4]; size_t xoff[4];
#pragma unroll
    for (int m = 0; m < 4; ++m) {
        tok[m] = tokenId[rt * 64 + m * 16 + lidx];
        // PRE: padding tokens read the dedicated zero row (index S_); branch-free K-loop
        xoff[m] = (size_t)(tok[m] < 0 ? (PRE ? S_ : 0) : tok[m]) * H_;
    }
    const int ng   = blockIdx.y * 128 + wave * 16;
    const int ncol = ng + lidx;
    const size_t gO = ((size_t)e * TWO_I + ncol) * H_;        // gate row
    const size_t uO = gO + (size_t)I_ * H_;                    // matched up row

    v8f accG[4] = {}; v8f accU[4] = {};
    for (int k0 = 0; k0 < H_; k0 += 32) {
        Frag bg, bu;
        if constexpr (PRE) {
            ld8u(gbf + gO + k0 + hs,      bg.u);  ld8u(gbf + gO + k0 + 16 + hs, bg.u + 4);
            ld8u(gbf + uO + k0 + hs,      bu.u);  ld8u(gbf + uO + k0 + 16 + hs, bu.u + 4);
        } else {
            cvt8u(g32 + gO + k0 + hs,      bg.u); cvt8u(g32 + gO + k0 + 16 + hs, bg.u + 4);
            cvt8u(g32 + uO + k0 + hs,      bu.u); cvt8u(g32 + uO + k0 + 16 + hs, bu.u + 4);
        }
#pragma unroll
        for (int m = 0; m < 4; ++m) {
            Frag a;
            if constexpr (PRE) {
                ld8u(xbf + xoff[m] + k0 + hs,      a.u);
                ld8u(xbf + xoff[m] + k0 + 16 + hs, a.u + 4);
            } else {
                if (tok[m] >= 0) {
                    cvt8u(x32 + xoff[m] + k0 + hs,      a.u);
                    cvt8u(x32 + xoff[m] + k0 + 16 + hs, a.u + 4);
                } else {
#pragma unroll
                    for (int j = 0; j < 8; ++j) a.u[j] = 0;
                }
            }
            accG[m] = __builtin_amdgcn_wmma_f32_16x16x32_bf16(false, a.v, false, bg.v, (short)0, accG[m], false, false);
            accU[m] = __builtin_amdgcn_wmma_f32_16x16x32_bf16(false, a.v, false, bu.v, (short)0, accU[m], false, false);
        }
    }
#pragma unroll
    for (int m = 0; m < 4; ++m) {
#pragma unroll
        for (int v = 0; v < 8; ++v) {
            const int M = v + hs;
            const float g = accG[m][v];
            const float u = accU[m][v];
            const float hval = (g / (1.0f + __expf(-g))) * u;
            hbuf[(size_t)(rt * 64 + m * 16 + M) * I_ + ncol] = f2bfu(hval);
        }
    }
}

// ---------------- phase 2: out[token] += w * (h @ Wd^T) ----------------
// grid = (ROWTILES, H/128), block = 256 (8 waves), 4 M-subtiles per wave

template<bool PRE>
__global__ void moe_down(const unsigned short* __restrict__ hbuf,
                         const float* __restrict__ d32, const unsigned short* __restrict__ dbf,
                         const int* __restrict__ rowExpert, const int* __restrict__ tokenId,
                         const float* __restrict__ wpair, float* __restrict__ out) {
    __shared__ int   stok[64];
    __shared__ float sw[64];
    if (threadIdx.x < 64) {
        stok[threadIdx.x] = tokenId[blockIdx.x * 64 + threadIdx.x];
        sw[threadIdx.x]   = wpair[blockIdx.x * 64 + threadIdx.x];
    }
    __syncthreads();

    const int wave = threadIdx.x >> 5;
    const int lane = threadIdx.x & 31;
    const int rt   = blockIdx.x;
    const int e    = rowExpert[rt];
    const int hs   = (lane < 16) ? 0 : 8;
    const int lidx = lane & 15;

    size_t hoff[4];
#pragma unroll
    for (int m = 0; m < 4; ++m)
        hoff[m] = (size_t)(rt * 64 + m * 16 + lidx) * I_;

    const int colbase = blockIdx.y * 128 + wave * 16;
    const int ncol    = colbase + lidx;
    const size_t wO   = ((size_t)e * H_ + ncol) * I_;

    v8f acc[4] = {};
    for (int k0 = 0; k0 < I_; k0 += 32) {
        Frag b;
        if constexpr (PRE) {
            ld8u(dbf + wO + k0 + hs,      b.u);
            ld8u(dbf + wO + k0 + 16 + hs, b.u + 4);
        } else {
            cvt8u(d32 + wO + k0 + hs,      b.u);
            cvt8u(d32 + wO + k0 + 16 + hs, b.u + 4);
        }
#pragma unroll
        for (int m = 0; m < 4; ++m) {
            Frag a;
            ld8u(hbuf + hoff[m] + k0 + hs,      a.u);
            ld8u(hbuf + hoff[m] + k0 + 16 + hs, a.u + 4);
            acc[m] = __builtin_amdgcn_wmma_f32_16x16x32_bf16(false, a.v, false, b.v, (short)0, acc[m], false, false);
        }
    }
#pragma unroll
    for (int m = 0; m < 4; ++m) {
#pragma unroll
        for (int v = 0; v < 8; ++v) {
            const int M = m * 16 + v + hs;
            const int tok = stok[M];
            const float w = sw[M];
            if (tok >= 0 && w != 0.f)
                atomicAdd(out + (size_t)tok * H_ + ncol, w * acc[m][v]);
        }
    }
}

// ---------------- launcher ----------------

extern "C" void kernel_launch(void* const* d_in, const int* in_sizes, int n_in,
                              void* d_out, int out_size, void* d_ws, size_t ws_size,
                              hipStream_t stream) {
    const float* x   = (const float*)d_in[0];   // (S,H) fp32
    const int*   idx = (const int*)  d_in[1];   // (S,K) int32
    const float* wts = (const float*)d_in[2];   // (S,K) fp32
    const float* gup = (const float*)d_in[3];   // (E,2I,H) fp32
    const float* dwn = (const float*)d_in[4];   // (E,H,I) fp32
    float* out = (float*)d_out;

    constexpr size_t HBUF_E = (size_t)ROWCAP * I_;          // bf16 elements
    constexpr size_t XBF_E  = (size_t)(S_ + 1) * H_;        // +1 zero row
    constexpr size_t GUP_E  = (size_t)E_ * TWO_I * H_;
    constexpr size_t DWN_E  = (size_t)E_ * H_ * I_;
    constexpr size_t META_B = 64 * 1024;
    constexpr size_t NEED_PRE = (HBUF_E + XBF_E + GUP_E + DWN_E) * 2 + META_B;

    unsigned short* hbuf = (unsigned short*)d_ws;
    const bool pre = (ws_size >= NEED_PRE);

    unsigned short *xbf = nullptr, *gbf = nullptr, *dbf = nullptr;
    unsigned int* xz;
    int* meta;
    if (pre) {
        xbf  = hbuf + HBUF_E;
        gbf  = xbf + XBF_E;
        dbf  = gbf + GUP_E;
        xz   = (unsigned int*)(xbf + (size_t)S_ * H_);       // zero row = xbf row S_
        meta = (int*)(dbf + DWN_E);
    } else {
        xz   = (unsigned int*)(hbuf + HBUF_E);               // 4KB dummy zone
        meta = (int*)((char*)(hbuf + HBUF_E) + 4096);
    }
    int* cnt       = meta;             // 8
    int* cursor    = meta + 8;         // 8
    int* off       = meta + 16;        // 9
    int* rowExpert = meta + 32;        // 72
    int* tokenId   = meta + 112;       // 4608
    float* wpair   = (float*)(meta + 112 + ROWCAP); // 4608

    moe_zero   <<<(S_ * H_ + 255) / 256, 256, 0, stream>>>(out, meta, xz);
    moe_count  <<<(NPAIR + 255) / 256,   256, 0, stream>>>(idx, cnt);
    moe_offsets<<<1,                     256, 0, stream>>>(cnt, off, rowExpert, tokenId, wpair);
    moe_scatter<<<(NPAIR + 255) / 256,   256, 0, stream>>>(idx, wts, cursor, off, tokenId, wpair);

    if (pre) {
        moe_cvt<<<(int)((S_ * (size_t)H_ / 8 + 255) / 256), 256, 0, stream>>>(x,   xbf, (int)(S_ * (size_t)H_ / 8));
        moe_cvt<<<(int)((GUP_E / 8 + 255) / 256),           256, 0, stream>>>(gup, gbf, (int)(GUP_E / 8));
        moe_cvt<<<(int)((DWN_E / 8 + 255) / 256),           256, 0, stream>>>(dwn, dbf, (int)(DWN_E / 8));
        moe_gateup<true><<<dim3(ROWTILES, I_ / 128), 256, 0, stream>>>(x, xbf, gup, gbf, rowExpert, tokenId, hbuf);
        moe_down<true>  <<<dim3(ROWTILES, H_ / 128), 256, 0, stream>>>(hbuf, dwn, dbf, rowExpert, tokenId, wpair, out);
    } else {
        moe_gateup<false><<<dim3(ROWTILES, I_ / 128), 256, 0, stream>>>(x, xbf, gup, gbf, rowExpert, tokenId, hbuf);
        moe_down<false>  <<<dim3(ROWTILES, H_ / 128), 256, 0, stream>>>(hbuf, dwn, dbf, rowExpert, tokenId, wpair, out);
    }
}
